// SingleStreamBlock_10831907520736
// MI455X (gfx1250) — matrix operand order repaired and checked by reading, not compiled, not run on hardware
//
#include <hip/hip_runtime.h>
#include <cstdint>

#define HIDDEN   3072
#define HEADS    24
#define HEAD_DIM 128
#define MLPH     12288
#define SEQ      2304
#define W1_N     (3 * HIDDEN + MLPH)   // 21504
#define QKV_N    (3 * HIDDEN)          // 9216
#define W2_K     (HIDDEN + MLPH)       // 15360
#define KTILES   (SEQ / 64)            // 36
#define EPSF     1e-6f

typedef __attribute__((ext_vector_type(16))) __bf16          v16bf;
typedef __attribute__((ext_vector_type(16))) unsigned short  v16u16;
typedef __attribute__((ext_vector_type(8)))  float           v8f;

// ---------------------------------------------------------------- bf16 helpers
__device__ __forceinline__ unsigned short f2bf(float f) {
  unsigned int u = __builtin_bit_cast(unsigned int, f);
  u += 0x7FFFu + ((u >> 16) & 1u);            // round-to-nearest-even
  return (unsigned short)(u >> 16);
}

__device__ __forceinline__ v8f wmma_bf16(v16bf a, v16bf b, v8f c) {
  // D = A(16x32 bf16) * B(32x16 bf16) + C(16x16 f32)
  return __builtin_amdgcn_wmma_f32_16x16x32_bf16(false, a, false, b, (short)0, c,
                                                 false, false);
}

// ------------------------------------------------- async global→LDS (CDNA5)
// 16 B per lane copied global→LDS, OFF added to both addresses ; tracked by ASYNCcnt
template <int OFF>
__device__ __forceinline__ void async_b128(unsigned lds, const void* g) {
  if constexpr (OFF == 0)
    asm volatile("global_load_async_to_lds_b128 %0, %1, off"
                 :: "v"(lds), "v"(g) : "memory");
  else
    asm volatile("global_load_async_to_lds_b128 %0, %1, off offset:%2"
                 :: "v"(lds), "v"(g), "i"(OFF) : "memory");
}
template <int N>
__device__ __forceinline__ void wait_async() {
  asm volatile("s_wait_asynccnt %0" :: "i"(N) : "memory");
}
__device__ __forceinline__ unsigned lds_addr(const void* p) {
  return (unsigned)(uintptr_t)p;   // generic LDS address truncates to LDS offset
}

// --------------------------------------------------------- fragment loaders
// A fragment, 16x32 bf16 from LDS. lane=g*16+m; element j: (M=m, K=16g+j)
__device__ __forceinline__ v16bf load_frag_a(const unsigned short* s, int ld,
                                             int row0, int k0, int lane) {
  const int g = lane >> 4, m = lane & 15;
  const unsigned short* p = s + (row0 + m) * ld + k0;
  v16u16 t;
#pragma unroll
  for (int j = 0; j < 16; ++j) {
    int kk = (g << 4) + j;
    t[j] = p[kk];
  }
  return __builtin_bit_cast(v16bf, t);
}
// B fragment where logical B = Xᵀ, X row-major [N][K] in LDS (Q·Kᵀ path)
__device__ __forceinline__ v16bf load_frag_bt(const unsigned short* s, int ld,
                                              int n0, int k0, int lane) {
  const int g = lane >> 4, m = lane & 15;
  const unsigned short* p = s + (n0 + m) * ld + k0 + (g << 4);
  v16u16 t;
#pragma unroll
  for (int j = 0; j < 16; ++j) t[j] = p[j];
  return __builtin_bit_cast(v16bf, t);
}
// B fragment from pre-swizzled global memory: frag base + lane*16, 32 contiguous B
__device__ __forceinline__ v16bf load_frag_g(const unsigned short* p) {
  struct U { uint4 a, b; } u;
  u.a = ((const uint4*)p)[0];
  u.b = ((const uint4*)p)[1];
  return __builtin_bit_cast(v16bf, u);
}

// --------------------- K0: f32 weights → bf16 fragment-swizzled global layout
// dst[((n>>4)*(K/32) + kblk)*512 + ((g<<4)|(n&15))*16 + j] = bf16(src[k*N+n])
//   with k = kblk*32 + g*16 + j
__global__ __launch_bounds__(256) void swizzle_w_kernel(
    const float* __restrict__ src, unsigned short* __restrict__ dst,
    int K, int N) {
  int tid = blockIdx.x * 256 + threadIdx.x;
  int total = (K >> 4) * N;
  if (tid >= total) return;
  int n = tid % N;
  int kh = tid / N;               // 0 .. K/16-1
  int kblk = kh >> 1, g = kh & 1;
  unsigned short tmp[16];
#pragma unroll
  for (int j = 0; j < 16; ++j)
    tmp[j] = f2bf(src[(size_t)(kblk * 32 + g * 16 + j) * N + n]);
  size_t base = ((size_t)(n >> 4) * (K >> 5) + kblk) * 512 +
                (size_t)(((g << 4) | (n & 15)) << 4);
  uint4* d = (uint4*)&dst[base];
  d[0] = *(uint4*)&tmp[0];
  d[1] = *(uint4*)&tmp[8];
}

// ------------------------------------------------------- K1: mod = silu(v)@Wm+b
__global__ __launch_bounds__(256) void mod_kernel(
    const float* __restrict__ vec, const float* __restrict__ w,
    const float* __restrict__ b, float* __restrict__ mod) {
  __shared__ float sv[HIDDEN];
  const int t = threadIdx.x;
  for (int i = t; i < HIDDEN; i += 256) {
    float v = vec[i];
    sv[i] = v / (1.f + __expf(-v));
  }
  __syncthreads();
  const int o = blockIdx.x * 256 + t;
  float acc = 0.f;
  for (int i = 0; i < HIDDEN; ++i) acc += sv[i] * w[i * QKV_N + o];
  mod[o] = acc + b[o];
}

// ------------------------------------------- K2: layernorm + modulate → bf16
__global__ __launch_bounds__(256) void ln_mod_kernel(
    const float* __restrict__ x, const float* __restrict__ mod,
    unsigned short* __restrict__ xmod) {
  __shared__ float red[256];
  const int s = blockIdx.x, t = threadIdx.x;
  const float* row = x + (size_t)s * HIDDEN;
  float sum = 0.f;
  for (int i = t; i < HIDDEN; i += 256) sum += row[i];
  red[t] = sum;
  __syncthreads();
  for (int off = 128; off > 0; off >>= 1) {
    if (t < off) red[t] += red[t + off];
    __syncthreads();
  }
  const float mu = red[0] * (1.f / HIDDEN);
  __syncthreads();
  float vs = 0.f;
  for (int i = t; i < HIDDEN; i += 256) {
    float d = row[i] - mu;
    vs += d * d;
  }
  red[t] = vs;
  __syncthreads();
  for (int off = 128; off > 0; off >>= 1) {
    if (t < off) red[t] += red[t + off];
    __syncthreads();
  }
  const float inv = rsqrtf(red[0] * (1.f / HIDDEN) + EPSF);
  for (int i = t; i < HIDDEN; i += 256) {
    float xm = (row[i] - mu) * inv;
    xmod[(size_t)s * HIDDEN + i] = f2bf(xm * (1.f + mod[HIDDEN + i]) + mod[i]);
  }
}

// -------------------------------------------------------------- GEMM1 (WMMA)
__global__ __launch_bounds__(256) void gemm1_kernel(
    const unsigned short* __restrict__ A,    // 2304 x 3072 bf16 row-major
    const unsigned short* __restrict__ Bsw,  // w1 fragment-swizzled bf16
    const float* __restrict__ bias,          // 21504
    float* __restrict__ qkv,                 // 2304 x 9216 f32
    unsigned short* __restrict__ mlp) {      // 2304 x 12288 gelu bf16
  const int K = HIDDEN, KB = K >> 5;         // 96 k-blocks
  __shared__ unsigned short sA[2][128 * 40];
  const int bn = blockIdx.x, bm = blockIdx.y;
  const int tid = threadIdx.x, lane = tid & 31, wave = tid >> 5;
  const int wm = (wave >> 2) * 64, wn = (wave & 3) * 32;
  v8f acc[4][2];
#pragma unroll
  for (int i = 0; i < 4; ++i)
#pragma unroll
    for (int j = 0; j < 2; ++j) acc[i][j] = (v8f)0.f;

  const int arow = tid >> 1, acol = (tid & 1) << 4;
  const unsigned short* Ag = A + (size_t)(bm * 128 + arow) * K + acol;
  const unsigned ldsA[2] = {lds_addr(&sA[0][arow * 40 + acol]),
                            lds_addr(&sA[1][arow * 40 + acol])};
  const unsigned short* B0 =
      Bsw + (size_t)(bn * 8 + (wave & 3) * 2) * KB * 512 + lane * 16;
  const unsigned short* B1 = B0 + (size_t)KB * 512;

  // prologue: tile 0 → buffer 0
  async_b128<0>(ldsA[0], Ag);
  async_b128<16>(ldsA[0], Ag);

  for (int kb = 0; kb < KB; ++kb) {
    __syncthreads();                 // everyone done with buffer (kb+1)&1
    const bool more = (kb + 1 < KB);
    if (more) {                      // issue tile kb+1 while computing tile kb
      const unsigned short* An = Ag + 32;
      unsigned dst = ldsA[(kb + 1) & 1];
      async_b128<0>(dst, An);
      async_b128<16>(dst, An);
      __builtin_prefetch(An + 32, 0, 3);
    }
    v16bf bf0 = load_frag_g(B0);
    v16bf bf1 = load_frag_g(B1);
    B0 += 512; B1 += 512;
    if (more) wait_async<2>();       // oldest 2 (tile kb) complete
    else      wait_async<0>();
    __syncthreads();                 // tile kb visible to all waves
    const unsigned short* sAc = sA[kb & 1];
#pragma unroll
    for (int im = 0; im < 4; ++im) {
      v16bf af = load_frag_a(sAc, 40, wm + im * 16, 0, lane);
      acc[im][0] = wmma_bf16(af, bf0, acc[im][0]);
      acc[im][1] = wmma_bf16(af, bf1, acc[im][1]);
    }
    Ag += 32;
  }
  const int g = lane >> 4, m = lane & 15;
  const bool is_qkv = (bn * 128) < QKV_N;    // uniform (9216 % 128 == 0)
#pragma unroll
  for (int im = 0; im < 4; ++im)
#pragma unroll
    for (int in = 0; in < 2; ++in)
#pragma unroll
      for (int r = 0; r < 8; ++r) {
        int gr = bm * 128 + wm + im * 16 + r + (g << 3);
        int gc = bn * 128 + wn + in * 16 + m;
        float v = acc[im][in][r] + bias[gc];
        if (is_qkv) {
          qkv[(size_t)gr * QKV_N + gc] = v;
        } else {
          float x3 = v * v * v;
          float gl = 0.5f * v *
                     (1.f + tanhf(0.7978845608028654f * (v + 0.044715f * x3)));
          mlp[(size_t)gr * MLPH + (gc - QKV_N)] = f2bf(gl);
        }
      }
}

// ----------------------------- K4: per-head RMSNorm + RoPE → bf16 Q/K/Vswz
__global__ __launch_bounds__(128) void qkv_post_kernel(
    const float* __restrict__ qkv, const float* __restrict__ qw,
    const float* __restrict__ kw, const float* __restrict__ cosb,
    const float* __restrict__ sinb, const int* __restrict__ txt_p,
    unsigned short* __restrict__ Qo, unsigned short* __restrict__ Ko,
    unsigned short* __restrict__ Vsw) {
  const int s = blockIdx.x, h = blockIdx.y, d = threadIdx.x;
  const int img = SEQ - txt_p[0];
  __shared__ float aq[128], ak[128], qs[128], ks[128];
  const float* base = qkv + (size_t)s * QKV_N + h * HEAD_DIM + d;
  float q = base[0], k = base[HIDDEN], v = base[2 * HIDDEN];
  aq[d] = q * q;
  ak[d] = k * k;
  __syncthreads();
  for (int off = 64; off > 0; off >>= 1) {
    if (d < off) { aq[d] += aq[d + off]; ak[d] += ak[d + off]; }
    __syncthreads();
  }
  const float rq = rsqrtf(aq[0] * (1.f / HEAD_DIM) + EPSF);
  const float rk = rsqrtf(ak[0] * (1.f / HEAD_DIM) + EPSF);
  qs[d] = q * rq * qw[d];
  ks[d] = k * rk * kw[d];
  __syncthreads();
  float qo = qs[d], ko = ks[d];
  if (s < img) {
    float c = cosb[(size_t)s * HEAD_DIM + d];
    float sn = sinb[(size_t)s * HEAD_DIM + d];
    float qp = qs[d ^ 1], kp = ks[d ^ 1];
    if ((d & 1) == 0) { qo = qo * c - qp * sn; ko = ko * c - kp * sn; }
    else              { qo = qo * c + qp * sn; ko = ko * c + kp * sn; }
  }
  const size_t o = (size_t)h * SEQ * HEAD_DIM + (size_t)s * HEAD_DIM + d;
  Qo[o] = f2bf(qo);
  Ko[o] = f2bf(ko);
  // V in WMMA B-fragment layout: per (h, key-tile kt, k-chunk kk, d-tile)
  {
    int kt = s >> 6, loc = s & 63, kk = loc >> 5, kr = loc & 31;
    int gg = kr >> 4, j = kr & 15;
    int dtile = d >> 4, lo = d & 15;
    size_t addr = ((((size_t)h * KTILES + kt) * 2 + kk) * 8 + dtile) * 512 +
                  (size_t)(((gg << 4) | lo) << 4) + j;
    Vsw[addr] = f2bf(v);
  }
}

// ---------------------------------------------- K5: flash attention (WMMA)
__global__ __launch_bounds__(256) void attn_kernel(
    const unsigned short* __restrict__ Q, const unsigned short* __restrict__ Kb,
    const unsigned short* __restrict__ Vsw, unsigned short* __restrict__ O) {
  __shared__ unsigned short sQ[64 * 136];
  __shared__ unsigned short sK[64 * 136];
  __shared__ float          sS[64 * 68];
  __shared__ unsigned short sP[64 * 72];
  __shared__ float rmax[64], rsum[64], ralpha[64];

  const int qt = blockIdx.x, h = blockIdx.y;
  const int tid = threadIdx.x, lane = tid & 31, wave = tid >> 5;
  const int wm = (wave >> 1) * 16;   // 16 query rows per wave
  const int wn = wave & 1;           // S: 32 cols/wave ; O: 64 cols/wave
  const int lr = tid >> 2, lc = (tid & 3) << 5;

  const unsigned short* Qg = Q + (size_t)h * SEQ * HEAD_DIM;
  const unsigned short* Kg = Kb + (size_t)h * SEQ * HEAD_DIM + (size_t)lr * HEAD_DIM + lc;
  const unsigned short* Vh = Vsw + (size_t)h * KTILES * 2 * 8 * 512 + lane * 16;
  const unsigned dk_lds = lds_addr(&sK[lr * 136 + lc]);

  {  // async load Q tile (64 x 128)
    const unsigned short* qg = Qg + (size_t)(qt * 64 + lr) * HEAD_DIM + lc;
    const unsigned dq = lds_addr(&sQ[lr * 136 + lc]);
    async_b128<0>(dq, qg); async_b128<16>(dq, qg);
    async_b128<32>(dq, qg); async_b128<48>(dq, qg);
  }
  {  // async load K tile 0
    async_b128<0>(dk_lds, Kg); async_b128<16>(dk_lds, Kg);
    async_b128<32>(dk_lds, Kg); async_b128<48>(dk_lds, Kg);
  }
  if (tid < 64) { rmax[tid] = -1e30f; rsum[tid] = 0.f; }

  v8f o_acc[4];
#pragma unroll
  for (int i = 0; i < 4; ++i) o_acc[i] = (v8f)0.f;

  const float sm = 0.08838834764831845f;  // 1/sqrt(128)
  const int g = lane >> 4, m = lane & 15;

  for (int kt = 0; kt < KTILES; ++kt) {
    wait_async<0>();                 // K(kt) (and Q on kt==0) landed
    __syncthreads();

    // S = Q · Kᵀ
    v8f s_acc[2];
    s_acc[0] = (v8f)0.f; s_acc[1] = (v8f)0.f;
#pragma unroll
    for (int dk = 0; dk < 4; ++dk) {
      v16bf a = load_frag_a(sQ, 136, wm, dk * 32, lane);
      v16bf b0 = load_frag_bt(sK, 136, wn * 32, dk * 32, lane);
      v16bf b1 = load_frag_bt(sK, 136, wn * 32 + 16, dk * 32, lane);
      s_acc[0] = wmma_bf16(a, b0, s_acc[0]);
      s_acc[1] = wmma_bf16(a, b1, s_acc[1]);
    }
#pragma unroll
    for (int in = 0; in < 2; ++in)
#pragma unroll
      for (int r = 0; r < 8; ++r)
        sS[(wm + r + (g << 3)) * 68 + wn * 32 + in * 16 + m] = s_acc[in][r] * sm;
    __syncthreads();                 // sS ready; all waves done reading sK

    if (kt + 1 < KTILES) {           // prefetch next K tile behind softmax+PV
      const unsigned short* kg = Kg + (size_t)(kt + 1) * 64 * HEAD_DIM;
      async_b128<0>(dk_lds, kg); async_b128<16>(dk_lds, kg);
      async_b128<32>(dk_lds, kg); async_b128<48>(dk_lds, kg);
    }
    if (tid < 64) {  // online softmax per query row
      float mo = rmax[tid];
      float mx = mo;
      const float* row = &sS[tid * 68];
      for (int j = 0; j < 64; ++j) mx = fmaxf(mx, row[j]);
      float al = __expf(mo - mx);
      float acc = rsum[tid] * al;
      for (int j = 0; j < 64; ++j) {
        float e = __expf(row[j] - mx);
        sP[tid * 72 + j] = f2bf(e);
        acc += e;
      }
      rmax[tid] = mx; rsum[tid] = acc; ralpha[tid] = al;
    }
    __syncthreads();                 // sP/stats ready

    // rescale running O, then O += P · V  (V fragments direct from global/L2)
#pragma unroll
    for (int r = 0; r < 8; ++r) {
      float al = ralpha[wm + r + (g << 3)];
#pragma unroll
      for (int in = 0; in < 4; ++in) o_acc[in][r] *= al;
    }
#pragma unroll
    for (int kk = 0; kk < 2; ++kk) {
      v16bf a = load_frag_a(sP, 72, wm, kk * 32, lane);
#pragma unroll
      for (int in = 0; in < 4; ++in) {
        v16bf b = load_frag_g(Vh + ((size_t)(kt * 2 + kk) * 8 + wn * 4 + in) * 512);
        o_acc[in] = wmma_bf16(a, b, o_acc[in]);
      }
    }
  }
  __syncthreads();

  // epilogue: normalize, store bf16 into [s][h*128+d]
#pragma unroll
  for (int r = 0; r < 8; ++r) {
    int row = wm + r + (g << 3);
    float inv = 1.f / rsum[row];
#pragma unroll
    for (int in = 0; in < 4; ++in) {
      int s = qt * 64 + row;
      int col = wn * 64 + in * 16 + m;
      O[(size_t)s * HIDDEN + h * HEAD_DIM + col] = f2bf(o_acc[in][r] * inv);
    }
  }
}

// -------------------------------------------------------------- GEMM2 (WMMA)
__global__ __launch_bounds__(256) void gemm2_kernel(
    const unsigned short* __restrict__ attn,  // 2304 x 3072 bf16
    const unsigned short* __restrict__ mlp,   // 2304 x 12288 bf16
    const unsigned short* __restrict__ Bsw,   // w2 fragment-swizzled bf16
    const float* __restrict__ b2, const float* __restrict__ gate,
    const float* __restrict__ x, float* __restrict__ out) {
  const int K = W2_K, KB = K >> 5;            // 480 k-blocks
  __shared__ unsigned short sA[2][128 * 40];
  const int bn = blockIdx.x, bm = blockIdx.y;
  const int tid = threadIdx.x, lane = tid & 31, wave = tid >> 5;
  const int wm = (wave >> 2) * 64, wn = (wave & 3) * 32;
  v8f acc[4][2];
#pragma unroll
  for (int i = 0; i < 4; ++i)
#pragma unroll
    for (int j = 0; j < 2; ++j) acc[i][j] = (v8f)0.f;

  const int arow = tid >> 1, acol = (tid & 1) << 4;
  const unsigned ldsA[2] = {lds_addr(&sA[0][arow * 40 + acol]),
                            lds_addr(&sA[1][arow * 40 + acol])};
  const unsigned short* Arow_attn = attn + (size_t)(bm * 128 + arow) * HIDDEN + acol;
  const unsigned short* Arow_mlp  = mlp + (size_t)(bm * 128 + arow) * MLPH + acol;
  auto aptr = [&](int kb) -> const unsigned short* {
    int k0 = kb * 32;
    return (k0 < HIDDEN) ? (Arow_attn + k0) : (Arow_mlp + (k0 - HIDDEN));
  };
  const unsigned short* B0 =
      Bsw + (size_t)(bn * 8 + (wave & 3) * 2) * KB * 512 + lane * 16;
  const unsigned short* B1 = B0 + (size_t)KB * 512;

  {  // prologue
    const unsigned short* Ap = aptr(0);
    async_b128<0>(ldsA[0], Ap);
    async_b128<16>(ldsA[0], Ap);
  }
  for (int kb = 0; kb < KB; ++kb) {
    __syncthreads();
    const bool more = (kb + 1 < KB);
    if (more) {
      const unsigned short* An = aptr(kb + 1);
      unsigned dst = ldsA[(kb + 1) & 1];
      async_b128<0>(dst, An);
      async_b128<16>(dst, An);
    }
    v16bf bf0 = load_frag_g(B0);
    v16bf bf1 = load_frag_g(B1);
    B0 += 512; B1 += 512;
    if (more) wait_async<2>();
    else      wait_async<0>();
    __syncthreads();
    const unsigned short* sAc = sA[kb & 1];
#pragma unroll
    for (int im = 0; im < 4; ++im) {
      v16bf af = load_frag_a(sAc, 40, wm + im * 16, 0, lane);
      acc[im][0] = wmma_bf16(af, bf0, acc[im][0]);
      acc[im][1] = wmma_bf16(af, bf1, acc[im][1]);
    }
  }
  const int g = lane >> 4, m = lane & 15;
#pragma unroll
  for (int im = 0; im < 4; ++im)
#pragma unroll
    for (int in = 0; in < 2; ++in)
#pragma unroll
      for (int r = 0; r < 8; ++r) {
        int gr = bm * 128 + wm + im * 16 + r + (g << 3);
        int gc = bn * 128 + wn + in * 16 + m;
        float v = acc[im][in][r] + b2[gc];
        size_t idx = (size_t)gr * HIDDEN + gc;
        out[idx] = x[idx] + gate[gc] * v;
      }
}

// ------------------------------------------------------------------- launcher
extern "C" void kernel_launch(void* const* d_in, const int* in_sizes, int n_in,
                              void* d_out, int out_size, void* d_ws, size_t ws_size,
                              hipStream_t stream) {
  (void)in_sizes; (void)n_in; (void)out_size; (void)ws_size;
  const float* x     = (const float*)d_in[0];
  const float* vec   = (const float*)d_in[1];
  const float* w_mod = (const float*)d_in[2];
  const float* b_mod = (const float*)d_in[3];
  const float* w1    = (const float*)d_in[4];
  const float* b1    = (const float*)d_in[5];
  const float* w2    = (const float*)d_in[6];
  const float* b2    = (const float*)d_in[7];
  const float* qw    = (const float*)d_in[8];
  const float* kw    = (const float*)d_in[9];
  const float* cosb  = (const float*)d_in[10];
  const float* sinb  = (const float*)d_in[11];
  const int*   txt   = (const int*)d_in[12];
  float* out = (float*)d_out;

  char* ws = (char*)d_ws;
  size_t off = 0;
  auto alloc = [&](size_t bytes) -> char* {
    char* p = ws + off;
    off = (off + bytes + 255) & ~(size_t)255;
    return p;
  };
  float*          modb = (float*)alloc((size_t)QKV_N * 4);
  unsigned short* xmod = (unsigned short*)alloc((size_t)SEQ * HIDDEN * 2);
  unsigned short* w1sw = (unsigned short*)alloc((size_t)HIDDEN * W1_N * 2);
  unsigned short* w2sw = (unsigned short*)alloc((size_t)W2_K * HIDDEN * 2);
  float*          qkvf = (float*)alloc((size_t)SEQ * QKV_N * 4);
  unsigned short* mlpb = (unsigned short*)alloc((size_t)SEQ * MLPH * 2);
  unsigned short* Qb   = (unsigned short*)alloc((size_t)SEQ * HIDDEN * 2);
  unsigned short* Kbuf = (unsigned short*)alloc((size_t)SEQ * HIDDEN * 2);
  unsigned short* Vsw  = (unsigned short*)alloc((size_t)SEQ * HIDDEN * 2);
  unsigned short* Ab   = (unsigned short*)alloc((size_t)SEQ * HIDDEN * 2);

  // K0: weight conversion + fragment swizzle (bf16 w1 then lives in L2)
  {
    int tot = (HIDDEN >> 4) * W1_N;
    swizzle_w_kernel<<<(tot + 255) / 256, 256, 0, stream>>>(w1, w1sw, HIDDEN, W1_N);
    tot = (W2_K >> 4) * HIDDEN;
    swizzle_w_kernel<<<(tot + 255) / 256, 256, 0, stream>>>(w2, w2sw, W2_K, HIDDEN);
  }
  mod_kernel<<<QKV_N / 256, 256, 0, stream>>>(vec, w_mod, b_mod, modb);
  ln_mod_kernel<<<SEQ, 256, 0, stream>>>(x, modb, xmod);
  gemm1_kernel<<<dim3(W1_N / 128, SEQ / 128), 256, 0, stream>>>(xmod, w1sw, b1,
                                                                qkvf, mlpb);
  qkv_post_kernel<<<dim3(SEQ, HEADS), 128, 0, stream>>>(qkvf, qw, kw, cosb, sinb,
                                                        txt, Qb, Kbuf, Vsw);
  attn_kernel<<<dim3(SEQ / 64, HEADS), 256, 0, stream>>>(Qb, Kbuf, Vsw, Ab);
  gemm2_kernel<<<dim3(HIDDEN / 128, SEQ / 128), 256, 0, stream>>>(
      Ab, mlpb, w2sw, b2, modb + 2 * HIDDEN, x, out);
}
